// MultiHeadAttention_2190433321380
// MI455X (gfx1250) — matrix operand order repaired and checked by reading, not compiled, not run on hardware
//
#include <hip/hip_runtime.h>

// ---------------------------------------------------------------------------
// MHA forward for MI455X (gfx1250, wave32, WMMA f32_16x16x32_f16 + TDM)
// B=8, C=768, T=1024, H=12, D=64
// ---------------------------------------------------------------------------

#define NB 8
#define NC 768
#define NT 1024
#define NH 12
#define ND 64

typedef __attribute__((ext_vector_type(16))) _Float16 v16h;
typedef __attribute__((ext_vector_type(8)))  _Float16 v8h;
typedef __attribute__((ext_vector_type(8)))  float    v8f;
typedef __attribute__((ext_vector_type(4)))  unsigned int u32x4;
typedef __attribute__((ext_vector_type(8)))  int          i32x8;
typedef __attribute__((ext_vector_type(4)))  int          i32x4;

#ifndef HAVE_TDM
#  if defined(__has_builtin)
#    if __has_builtin(__builtin_amdgcn_tensor_load_to_lds)
#      define HAVE_TDM 1
#    else
#      define HAVE_TDM 0
#    endif
#  else
#    define HAVE_TDM 0
#  endif
#endif

static constexpr float ATT_SCALE = 0.125f; // 64^-0.5

static __device__ __forceinline__ v16h cat8(v8h a, v8h b) {
  return __builtin_shufflevector(a, b, 0,1,2,3,4,5,6,7,8,9,10,11,12,13,14,15);
}
static __device__ __forceinline__ v8f wmma16(v16h a, v16h b, v8f c) {
  // (neg_a, A, neg_b, B, c_mod, C, reuse_a, reuse_b)
  return __builtin_amdgcn_wmma_f32_16x16x32_f16(false, a, false, b, (short)0, c,
                                                false, false);
}
// Reduce across a 16-lane half (rows of the 16x16 C/D tile live in one half).
static __device__ __forceinline__ float halfmax(float v) {
  v = fmaxf(v, __shfl_xor(v, 1, 32));
  v = fmaxf(v, __shfl_xor(v, 2, 32));
  v = fmaxf(v, __shfl_xor(v, 4, 32));
  v = fmaxf(v, __shfl_xor(v, 8, 32));
  return v;
}
static __device__ __forceinline__ float halfsum(float v) {
  v += __shfl_xor(v, 1, 32);
  v += __shfl_xor(v, 2, 32);
  v += __shfl_xor(v, 4, 32);
  v += __shfl_xor(v, 8, 32);
  return v;
}

// ---------------------------------------------------------------------------
// Projection GEMM:  Y[o,t] = sum_i W[o,i] * X[i,t] + bias[o]
// Block = 128 threads (4 waves); wave w: rows [m0, m0+16), cols [n0, n0+64).
// IN_F16=false : X is f32 [B,C,T]   (QKV projections, input x or c)
// IN_F16=true  : X is f16 [B,H,T,D] read head-merged (output projection)
// OUT_SPLIT=true : write f16 [B,H,T,D]; else f32 [B,C,T] (d_out)
// ---------------------------------------------------------------------------
template<bool IN_F16, bool OUT_SPLIT>
__global__ __launch_bounds__(128)
void proj_kernel(const float* __restrict__ W, const float* __restrict__ bias,
                 const float* __restrict__ Xf, const _Float16* __restrict__ Xh,
                 _Float16* __restrict__ Yh, float* __restrict__ Yf)
{
  __shared__ _Float16 ldsB[64 * 32]; // transposed activation tile: [n][k]

  const int tid  = threadIdx.x;
  const int wv   = tid >> 5;
  const int lane = tid & 31;
  const int half = lane >> 4;
  const int l16  = lane & 15;
  const int b    = blockIdx.z;
  const int m0   = blockIdx.y * 64 + wv * 16;
  const int n0   = blockIdx.x * 64;
  const int kbA  = half ? 8 : 0;   // A-frag K runs: [kbA,kbA+8) and [kbA+16,kbA+24)
  const int kbB  = half ? 8 : 0;   // B-frag K runs: [kbB,kbB+8) and [kbB+16,kbB+24)

  v8f acc[4] = {};

  for (int k0 = 0; k0 < NC; k0 += 32) {
    __syncthreads();
    if constexpr (!IN_F16) {
      // stage 32(k) x 64(n) f32 tile of X -> LDS f16 transposed [n][k]
      const int kr = tid >> 2;        // 0..31
      const int nb = (tid & 3) * 16;  // 0,16,32,48
      const float* src = Xf + ((size_t)b * NC + k0 + kr) * NT + n0 + nb;
      float4 f0 = *(const float4*)(src + 0);
      float4 f1 = *(const float4*)(src + 4);
      float4 f2 = *(const float4*)(src + 8);
      float4 f3 = *(const float4*)(src + 12);
      float fv[16] = {f0.x,f0.y,f0.z,f0.w, f1.x,f1.y,f1.z,f1.w,
                      f2.x,f2.y,f2.z,f2.w, f3.x,f3.y,f3.z,f3.w};
      #pragma unroll
      for (int j = 0; j < 16; ++j)
        ldsB[(nb + j) * 32 + kr] = (_Float16)fv[j];
    } else {
      // channel kk = h*64 + d; kk-range [k0+ks, +16) stays inside one head
      const int nr = tid >> 1;        // 0..63
      const int ks = (tid & 1) * 16;  // 0 or 16
      const int kk = k0 + ks;
      const _Float16* src =
          Xh + (((size_t)b * NH + (kk >> 6)) * NT + (n0 + nr)) * ND + (kk & 63);
      v8h a0 = *(const v8h*)(src);
      v8h a1 = *(const v8h*)(src + 8);
      *(v8h*)&ldsB[nr * 32 + ks + 0] = a0;
      *(v8h*)&ldsB[nr * 32 + ks + 8] = a1;
    }
    __syncthreads();

    // A fragment from W (row-major f32 [O,I]) with on-the-fly f32->f16 cvt
    const float* wrow = W + (size_t)(m0 + l16) * NC + k0 + kbA;
    float4 w0 = *(const float4*)(wrow + 0);
    float4 w1 = *(const float4*)(wrow + 4);
    float4 w2 = *(const float4*)(wrow + 16);
    float4 w3 = *(const float4*)(wrow + 20);
    v16h Af;
    Af[0]=(_Float16)w0.x;  Af[1]=(_Float16)w0.y;  Af[2]=(_Float16)w0.z;  Af[3]=(_Float16)w0.w;
    Af[4]=(_Float16)w1.x;  Af[5]=(_Float16)w1.y;  Af[6]=(_Float16)w1.z;  Af[7]=(_Float16)w1.w;
    Af[8]=(_Float16)w2.x;  Af[9]=(_Float16)w2.y;  Af[10]=(_Float16)w2.z; Af[11]=(_Float16)w2.w;
    Af[12]=(_Float16)w3.x; Af[13]=(_Float16)w3.y; Af[14]=(_Float16)w3.z; Af[15]=(_Float16)w3.w;

    #pragma unroll
    for (int nt = 0; nt < 4; ++nt) {
      const _Float16* bp = &ldsB[(nt * 16 + l16) * 32 + kbB];
      v16h Bf = cat8(*(const v8h*)bp, *(const v8h*)(bp + 16));
      acc[nt] = wmma16(Af, Bf, acc[nt]);
    }
  }

  // C/D layout: VGPR r -> row M = r + 8*half, col N = l16 within each 16-tile
  #pragma unroll
  for (int r = 0; r < 8; ++r) {
    const int o = m0 + r + half * 8;
    const float bo = bias[o];
    #pragma unroll
    for (int nt = 0; nt < 4; ++nt) {
      const int t = n0 + nt * 16 + l16;
      const float vout = acc[nt][r] + bo;
      if constexpr (OUT_SPLIT)
        Yh[(((size_t)b * NH + (o >> 6)) * NT + t) * ND + (o & 63)] = (_Float16)vout;
      else
        Yf[((size_t)b * NC + o) * NT + t] = vout;
    }
  }
}

// ---------------------------------------------------------------------------
// Flash attention: one wave = 16 query rows; block (4 waves) = 64 query rows,
// shares LDS-staged 32-wide K/V tiles. K tile is DMA'd by the Tensor Data
// Mover (2D tile 64(d) x 32(s), f16) when the toolchain exposes the builtin;
// V is staged transposed by VALU (TDM cannot transpose).
// ---------------------------------------------------------------------------
__global__ __launch_bounds__(128)
void flash_attn_kernel(const _Float16* __restrict__ Q, const _Float16* __restrict__ Km,
                       const _Float16* __restrict__ Vm, _Float16* __restrict__ Om)
{
  __shared__ _Float16 ldsK[32 * 64];    // [s][d]
  __shared__ _Float16 ldsV[64 * 32];    // [d][s]  (transposed for P*V B-frags)
  __shared__ _Float16 ldsP[4][16 * 32]; // per-wave P tile, [m][k]

  const int tid  = threadIdx.x;
  const int wv   = tid >> 5;
  const int lane = tid & 31;
  const int half = lane >> 4;
  const int l16  = lane & 15;
  const int b    = blockIdx.z;
  const int h    = blockIdx.y;
  const size_t base = ((size_t)b * NH + h) * NT * ND;
  const int tq0  = blockIdx.x * 64 + wv * 16;
  const int kbA  = half ? 16 : 0;

  // Q A-fragments: rows t, K-dim d (two frags cover d=0..63)
  v16h qA[2];
  #pragma unroll
  for (int f = 0; f < 2; ++f) {
    const _Float16* qp = Q + base + (size_t)(tq0 + l16) * ND + f * 32 + kbA;
    qA[f] = cat8(*(const v8h*)qp, *(const v8h*)(qp + 8));
  }

  v8f oacc[4] = {};
  float mrow[8], lrow[8];
  #pragma unroll
  for (int r = 0; r < 8; ++r) { mrow[r] = -1.0e30f; lrow[r] = 0.0f; }

  for (int s0 = 0; s0 < NT; s0 += 32) {
    __syncthreads();

#if HAVE_TDM
    if (wv == 0) {
      // --- Tensor Data Mover: DMA K tile [32 s][64 d] f16 into ldsK ---
      // D# per CDNA5 ISA (2D, groups 2/3 zero).
      const unsigned lds_off = (unsigned)(unsigned long long)(&ldsK[0]);
      const unsigned long long ga =
          (unsigned long long)(Km + base + (size_t)s0 * ND);
      u32x4 g0;
      g0[0] = 1u;                                        // count=1, user D#
      g0[1] = lds_off;                                   // lds_addr (bytes)
      g0[2] = (unsigned)(ga & 0xffffffffu);              // global_addr[31:0]
      g0[3] = ((unsigned)(ga >> 32) & 0x01ffffffu) | 0x80000000u; // [56:32]|type=2
      i32x8 g1;
      g1[0] = 0x00010000;                // data_size=1 (2B), wg_mask=0, no flags
      g1[1] = (int)(64u << 16);          // tensor_dim0 = 64  (d)
      g1[2] = (int)((unsigned)NT << 16); // tensor_dim1 = 1024 (s)
      g1[3] = (int)(64u << 16);          // tile_dim0 = 64
      g1[4] = 32;                        // tile_dim1 = 32, tile_dim2 = 0
      g1[5] = 64;                        // tensor_dim0_stride = 64 elems
      g1[6] = 0;                         // stride0 hi | stride1 lo16 (65536 -> 0)
      g1[7] = 1;                         // stride1[47:16] -> 64*1024 elems
      i32x4 z4 = {0, 0, 0, 0};
#if __clang_major__ >= 23
      i32x8 z8 = {0, 0, 0, 0, 0, 0, 0, 0};
      __builtin_amdgcn_tensor_load_to_lds(g0, g1, z4, z4, z8, 0);
#else
      __builtin_amdgcn_tensor_load_to_lds(g0, g1, z4, z4, 0);
#endif
    }
#endif
    {
      const int sr = tid >> 2;        // 0..31
      const int db = (tid & 3) * 16;  // 0,16,32,48
#if !HAVE_TDM
      const _Float16* kp = Km + base + (size_t)(s0 + sr) * ND + db;
      v8h ka = *(const v8h*)kp, kc = *(const v8h*)(kp + 8);
      *(v8h*)&ldsK[sr * 64 + db + 0] = ka;
      *(v8h*)&ldsK[sr * 64 + db + 8] = kc;
#endif
      const _Float16* vp = Vm + base + (size_t)(s0 + sr) * ND + db;
      v8h va = *(const v8h*)vp, vc = *(const v8h*)(vp + 8);
      #pragma unroll
      for (int j = 0; j < 8; ++j) {
        ldsV[(db + j) * 32 + sr]     = va[j];
        ldsV[(db + 8 + j) * 32 + sr] = vc[j];
      }
      if (s0 + 32 < NT) { // emits global_prefetch_b8 for next tiles
        __builtin_prefetch(Km + base + (size_t)(s0 + 32 + sr) * ND + db, 0, 1);
        __builtin_prefetch(Vm + base + (size_t)(s0 + 32 + sr) * ND + db, 0, 1);
      }
    }
#if HAVE_TDM
    if (wv == 0) __builtin_amdgcn_s_wait_tensorcnt(0); // TENSORcnt == 0
#endif
    __syncthreads();

    // scores S = Q * K^T for two 16-wide s tiles (K-dim d = 64 -> 2 wmma each)
    v8f S0 = {}, S1 = {};
    #pragma unroll
    for (int f = 0; f < 2; ++f) {
      const _Float16* bp0 = &ldsK[(0 * 16 + l16) * 64 + f * 32 + half * 16];
      v16h B0 = cat8(*(const v8h*)bp0, *(const v8h*)(bp0 + 8));
      S0 = wmma16(qA[f], B0, S0);
      const _Float16* bp1 = &ldsK[(1 * 16 + l16) * 64 + f * 32 + half * 16];
      v16h B1 = cat8(*(const v8h*)bp1, *(const v8h*)(bp1 + 8));
      S1 = wmma16(qA[f], B1, S1);
    }

    // online softmax, row r of accumulators == matrix row (r + 8*half)
    #pragma unroll
    for (int r = 0; r < 8; ++r) {
      float a0 = S0[r] * ATT_SCALE;
      float a1 = S1[r] * ATT_SCALE;
      float mx    = halfmax(fmaxf(a0, a1));
      float mnew  = fmaxf(mrow[r], mx);
      float alpha = __expf(mrow[r] - mnew);
      float p0 = __expf(a0 - mnew);
      float p1 = __expf(a1 - mnew);
      float rs = halfsum(p0 + p1);
      lrow[r] = lrow[r] * alpha + rs;
      mrow[r] = mnew;
      #pragma unroll
      for (int nt = 0; nt < 4; ++nt) oacc[nt][r] *= alpha;
      const int m = r + half * 8;                 // C-layout -> A-layout via LDS
      ldsP[wv][m * 32 + l16]      = (_Float16)p0; // k = l16       (s tile 0)
      ldsP[wv][m * 32 + 16 + l16] = (_Float16)p1; // k = 16 + l16  (s tile 1)
    }

    // O += P * V   (A-frag of P from per-wave LDS, B-frags of V transposed)
    const _Float16* pp = &ldsP[wv][l16 * 32 + kbA];
    v16h Pf = cat8(*(const v8h*)pp, *(const v8h*)(pp + 8));
    #pragma unroll
    for (int nt = 0; nt < 4; ++nt) {
      const _Float16* vb = &ldsV[(nt * 16 + l16) * 32 + half * 16];
      v16h Vf = cat8(*(const v8h*)vb, *(const v8h*)(vb + 8));
      oacc[nt] = wmma16(Pf, Vf, oacc[nt]);
    }
  }

  #pragma unroll
  for (int r = 0; r < 8; ++r) {
    const float inv = 1.0f / lrow[r];
    const int t = tq0 + r + half * 8;
    #pragma unroll
    for (int nt = 0; nt < 4; ++nt)
      Om[base + (size_t)t * ND + nt * 16 + l16] = (_Float16)(oacc[nt][r] * inv);
  }
}

// ---------------------------------------------------------------------------
extern "C" void kernel_launch(void* const* d_in, const int* in_sizes, int n_in,
                              void* d_out, int out_size, void* d_ws, size_t ws_size,
                              hipStream_t stream)
{
  const float* x  = (const float*)d_in[0];
  const float* c  = (const float*)d_in[1];
  const float* Wq = (const float*)d_in[2];
  const float* bq = (const float*)d_in[3];
  const float* Wk = (const float*)d_in[4];
  const float* bk = (const float*)d_in[5];
  const float* Wv = (const float*)d_in[6];
  const float* bv = (const float*)d_in[7];
  const float* Wo = (const float*)d_in[8];
  const float* bo = (const float*)d_in[9];
  float* out = (float*)d_out;

  const size_t per = (size_t)NB * NH * NT * ND; // 6,291,456 f16 elements each
  _Float16* Qw = (_Float16*)d_ws;
  _Float16* Kw = Qw + per;
  _Float16* Vw = Kw + per;
  _Float16* Ow = Vw + per; // total ws use: 4*per*2 = 48 MiB

  dim3 gp(NT / 64, NC / 64, NB); // 16 x 12 x 8 blocks, 128 threads
  dim3 ga(NT / 64, NH, NB);      // 16 x 12 x 8 blocks, 128 threads

  proj_kernel<false, true ><<<gp, 128, 0, stream>>>(Wq, bq, x, nullptr, Qw, nullptr);
  proj_kernel<false, true ><<<gp, 128, 0, stream>>>(Wk, bk, c, nullptr, Kw, nullptr);
  proj_kernel<false, true ><<<gp, 128, 0, stream>>>(Wv, bv, c, nullptr, Vw, nullptr);
  flash_attn_kernel<<<ga, 128, 0, stream>>>(Qw, Kw, Vw, Ow);
  proj_kernel<true,  false><<<gp, 128, 0, stream>>>(Wo, bo, nullptr, Ow, nullptr, out);

  (void)in_sizes; (void)n_in; (void)out_size; (void)ws_size;
}
